// WindowAttentionGlobal_34273839022291
// MI455X (gfx1250) — matrix operand order repaired and checked by reading, not compiled, not run on hardware
//
#include <hip/hip_runtime.h>
#include <hip/hip_bf16.h>

// ---------------- types ----------------
typedef _Float16 h8  __attribute__((ext_vector_type(8)));
typedef _Float16 h16 __attribute__((ext_vector_type(16)));
typedef float    f8  __attribute__((ext_vector_type(8)));

#if defined(__has_builtin)
#  if __has_builtin(__builtin_amdgcn_global_load_async_to_lds_b128)
#    define HAVE_ASYNC_LDS 1
#  else
#    define HAVE_ASYNC_LDS 0
#  endif
#else
#  define HAVE_ASYNC_LDS 0
#endif

#if HAVE_ASYNC_LDS
// Signature leaked by clang diagnostic: param0 = vector_size(16) int, addrspace(1);
// param1 = LDS side; params 2,3 = imm offset / cpol.
typedef int a_int4 __attribute__((vector_size(4 * sizeof(int))));
__device__ __forceinline__ void async_copy_b128(const float* g, float* l) {
    __builtin_amdgcn_global_load_async_to_lds_b128(
        (__attribute__((address_space(1))) a_int4*)g,
        (__attribute__((address_space(3))) a_int4*)l,
        0, 0);
}
#endif

__device__ __forceinline__ h16 cat16(h8 a, h8 b) {
    return __builtin_shufflevector(a, b, 0,1,2,3,4,5,6,7,8,9,10,11,12,13,14,15);
}
__device__ __forceinline__ f8 zero8() {
    f8 z;
#pragma unroll
    for (int i = 0; i < 8; ++i) z[i] = 0.f;
    return z;
}

// WMMA A operand (16x32 f16): lanes 0-15 hold row M=lane, K {0..7,16..23};
// lanes 16-31 hold row M=lane-16, K {8..15,24..31}.
__device__ __forceinline__ h16 load_a_lds(const _Float16* base, int stride,
                                          int row0, int kbase, int lane) {
    int r   = row0 + (lane & 15);
    int sel = (lane >> 4) & 1;
    const _Float16* p = base + r * stride + kbase + sel * 8;
    h8 lo = *(const h8*)p;
    h8 hi = *(const h8*)(p + 16);
    return cat16(lo, hi);
}
// A operand gathered from global f16 (row-clamped to 48: pad rows garbage-tolerant).
__device__ __forceinline__ h16 load_a_gf16(const _Float16* base, int stride,
                                           int row0, int kbase, int lane) {
    int r = row0 + (lane & 15); if (r > 48) r = 48;
    int sel = (lane >> 4) & 1;
    const _Float16* p = base + (size_t)r * stride + kbase + sel * 8;
    h8 lo = *(const h8*)p;
    h8 hi = *(const h8*)(p + 16);
    return cat16(lo, hi);
}
// A operand from global f32 with scale (fallback path).
__device__ __forceinline__ h16 load_a_gf32(const float* base, int stride,
                                           int row0, int kbase, int lane, float scl) {
    int r = row0 + (lane & 15); if (r > 48) r = 48;
    int sel = (lane >> 4) & 1;
    const float4* p = (const float4*)(base + (size_t)r * stride + kbase + sel * 8);
    float4 a = p[0], b = p[1], c = p[4], d = p[5];
    h16 v;
    v[0]=(_Float16)(a.x*scl);  v[1]=(_Float16)(a.y*scl);  v[2]=(_Float16)(a.z*scl);  v[3]=(_Float16)(a.w*scl);
    v[4]=(_Float16)(b.x*scl);  v[5]=(_Float16)(b.y*scl);  v[6]=(_Float16)(b.z*scl);  v[7]=(_Float16)(b.w*scl);
    v[8]=(_Float16)(c.x*scl);  v[9]=(_Float16)(c.y*scl);  v[10]=(_Float16)(c.z*scl); v[11]=(_Float16)(c.w*scl);
    v[12]=(_Float16)(d.x*scl); v[13]=(_Float16)(d.y*scl); v[14]=(_Float16)(d.z*scl); v[15]=(_Float16)(d.w*scl);
    return v;
}

// WMMA B operand (32x16 f16): lanes 0-15 col N=lane, K {0..7,16..23};
// lanes 16-31 col N=lane-16, K {8..15,24..31}. Storage: logical column n = contiguous row n.
__device__ __forceinline__ h16 load_b_lds(const _Float16* base, int stride,
                                          int col0, int kbase, int lane) {
    int n   = col0 + (lane & 15);
    int sel = (lane >> 4) & 1;
    const _Float16* p = base + n * stride + kbase + sel * 8;
    h8 lo = *(const h8*)p;
    h8 hi = *(const h8*)(p + 16);
    return cat16(lo, hi);
}
// B operand from pre-converted f16 weights in global (L2 resident).
__device__ __forceinline__ h16 load_b_gf16(const _Float16* base, int stride,
                                           int col0, int kbase, int lane) {
    int n   = col0 + (lane & 15);
    int sel = (lane >> 4) & 1;
    const h8* p = (const h8*)(base + (size_t)n * stride + kbase + sel * 8);
    return cat16(p[0], p[2]);
}
// B operand from f32 weights with conversion (fallback path).
__device__ __forceinline__ h16 load_b_gf32(const float* base, int stride,
                                           int col0, int kbase, int lane) {
    int n   = col0 + (lane & 15);
    int sel = (lane >> 4) & 1;
    const float4* p = (const float4*)(base + (size_t)n * stride + kbase + sel * 8);
    float4 a = p[0], b = p[1], c = p[4], d = p[5];
    h16 r;
    r[0]=(_Float16)a.x;  r[1]=(_Float16)a.y;  r[2]=(_Float16)a.z;  r[3]=(_Float16)a.w;
    r[4]=(_Float16)b.x;  r[5]=(_Float16)b.y;  r[6]=(_Float16)b.z;  r[7]=(_Float16)b.w;
    r[8]=(_Float16)c.x;  r[9]=(_Float16)c.y;  r[10]=(_Float16)c.z; r[11]=(_Float16)c.w;
    r[12]=(_Float16)d.x; r[13]=(_Float16)d.y; r[14]=(_Float16)d.z; r[15]=(_Float16)d.w;
    return r;
}

#define WMMA_F16(A, B, C) \
    __builtin_amdgcn_wmma_f32_16x16x32_f16(false, (A), false, (B), (short)0, (C), false, false)

// prep: one-shot f32->f16 of weights and pre-scaled global query into d_ws.
__global__ __launch_bounds__(256)
void wga_prep(const float* __restrict__ qg, const float* __restrict__ wqkv,
              const float* __restrict__ wproj,
              _Float16* __restrict__ qh, _Float16* __restrict__ wqkvh,
              _Float16* __restrict__ wprojh) {
    int i = blockIdx.x * 256 + threadIdx.x;
    const float scl = 0.1767766952966369f;  // 32^-0.5
    if (i < 301056) qh[i]     = (_Float16)(qg[i] * scl);
    if (i < 73728)  wqkvh[i]  = (_Float16)wqkv[i];
    if (i < 36864)  wprojh[i] = (_Float16)wproj[i];
}

// dims: B_=2048 windows, N=49(pad 64), H=6, D=32, C=192, 64 windows/image
template <bool WF16>
__global__ __launch_bounds__(256)
void wga_fused(const float* __restrict__ x,
               const float* __restrict__ qg,  const _Float16* __restrict__ qh,
               const float* __restrict__ wqkv, const _Float16* __restrict__ wqkvh,
               const float* __restrict__ bqkv,
               const float* __restrict__ btab,
               const float* __restrict__ wproj, const _Float16* __restrict__ wprojh,
               const float* __restrict__ bproj,
               float* __restrict__ out) {
    // sPool: phase A = f32 x staging (37.6KB), phase C = softmax probs (48KB)
    __shared__ __align__(16) unsigned char sPool[6 * 64 * 64 * 2];
    __shared__ __align__(16) _Float16 sXO[64 * 192];   // x(f16) then context
    __shared__ __align__(16) _Float16 sK [64 * 192];   // [n][h*32+d]
    __shared__ __align__(16) _Float16 sVt[192 * 64];   // [h*32+d][n]
    __shared__ __align__(16) float    sBias[169 * 6];
    _Float16* sP    = (_Float16*)sPool;
    float*    sXf32 = (float*)sPool;

    const int t     = threadIdx.x;
    const int lane  = t & 31;
    const int wv    = t >> 5;
    const int w     = blockIdx.x;
    const int img   = w >> 6;
    const int rhalf = ((lane >> 4) & 1) * 8;
    const int ncol  = lane & 15;

    // ---------------- Phase A: stage x tile + bias table ----------------
    const float* xw = x + (size_t)w * 49 * 192;
#if HAVE_ASYNC_LDS
    for (int i = t; i < 2352; i += 256) {  // 49*192/4 float4 chunks, HBM -> LDS direct
        async_copy_b128(xw + (size_t)i * 4, sXf32 + (size_t)i * 4);
    }
    for (int i = t; i < 169 * 6; i += 256) sBias[i] = btab[i];
#  if __has_builtin(__builtin_amdgcn_s_wait_asynccnt)
    __builtin_amdgcn_s_wait_asynccnt(0);
#  else
    asm volatile("s_wait_asynccnt 0" ::: "memory");
#  endif
    __syncthreads();
    for (int i = t; i < 64 * 192; i += 256) {
        int r = i / 192, c = i - r * 192;
        sXO[i] = (r < 49) ? (_Float16)sXf32[r * 192 + c] : (_Float16)0.f;
    }
#else
    for (int i = t; i < 169 * 6; i += 256) sBias[i] = btab[i];
    for (int i = t; i < 64 * 192; i += 256) {
        int r = i / 192, c = i - r * 192;
        sXO[i] = (r < 49) ? (_Float16)xw[r * 192 + c] : (_Float16)0.f;
    }
#endif
    __syncthreads();

    // ---------------- Phase B: KV projection kv = x @ Wqkv^T + b ----------------
    for (int ct = wv; ct < 24; ct += 8) {
        f8 acc[4];
#pragma unroll
        for (int mt = 0; mt < 4; ++mt) acc[mt] = zero8();
        for (int kk = 0; kk < 6; ++kk) {
            h16 bt;
            if constexpr (WF16) bt = load_b_gf16(wqkvh, 192, ct * 16, kk * 32, lane);
            else                bt = load_b_gf32(wqkv,  192, ct * 16, kk * 32, lane);
#pragma unroll
            for (int mt = 0; mt < 4; ++mt) {
                h16 at = load_a_lds(sXO, 192, mt * 16, kk * 32, lane);
                acc[mt] = WMMA_F16(at, bt, acc[mt]);
            }
        }
        const int colg  = ct * 16 + ncol;
        const float bias = bqkv[colg];
        if (ct < 12) {  // K -> sK[n][c] (row-strided scalar stores)
#pragma unroll
            for (int mt = 0; mt < 4; ++mt)
#pragma unroll
                for (int j = 0; j < 8; ++j)
                    sK[(mt * 16 + j + rhalf) * 192 + colg] =
                        (_Float16)(acc[mt][j] + bias);
        } else {        // V -> sVt[c][n] transposed: j runs contiguous -> b128 store
            const int vc = colg - 192;
#pragma unroll
            for (int mt = 0; mt < 4; ++mt) {
                h8 pk;
#pragma unroll
                for (int j = 0; j < 8; ++j) pk[j] = (_Float16)(acc[mt][j] + bias);
                *(h8*)&sVt[vc * 64 + mt * 16 + rhalf] = pk;
            }
        }
    }
    __syncthreads();

    // ---------------- Phase C: attention, one head per wave (waves 0..5) ----------------
    if (wv < 6) {
        const int h = wv;
        _Float16* sPh = sP + h * 64 * 64;
        const _Float16* qhb = qh + ((size_t)(img * 6 + h) * 49) * 32;
        const float*    qfb = qg + ((size_t)(img * 6 + h) * 49) * 32;
        for (int mt = 0; mt < 4; ++mt) {
            h16 qa;
            if constexpr (WF16) qa = load_a_gf16(qhb, 32, mt * 16, 0, lane);
            else                qa = load_a_gf32(qfb, 32, mt * 16, 0, lane, 0.1767766952966369f);
            f8 s[4];
#pragma unroll
            for (int nt = 0; nt < 4; ++nt) {
                h16 kb = load_b_lds(sK, 192, nt * 16, h * 32, lane);
                s[nt] = WMMA_F16(qa, kb, zero8());
            }
            // relative-position bias + pad-column mask
#pragma unroll
            for (int nt = 0; nt < 4; ++nt) {
                int n = nt * 16 + ncol;
#pragma unroll
                for (int j = 0; j < 8; ++j) {
                    int m = mt * 16 + j + rhalf;
                    float v = s[nt][j];
                    if (n >= 49) {
                        v = -1e30f;
                    } else if (m < 49) {
                        int im = m / 7, jm = m - im * 7;
                        int in = n / 7, jn = n - in * 7;
                        int idx = (im - in + 6) * 13 + (jm - jn + 6);
                        v += sBias[idx * 6 + h];
                    }
                    s[nt][j] = v;
                }
            }
            // softmax: 4 regs in-thread + 16-lane butterflies (rows live per half-wave)
#pragma unroll
            for (int j = 0; j < 8; ++j) {
                float mx = fmaxf(fmaxf(s[0][j], s[1][j]), fmaxf(s[2][j], s[3][j]));
#pragma unroll
                for (int o = 1; o < 16; o <<= 1) mx = fmaxf(mx, __shfl_xor(mx, o, 16));
                float e0 = __expf(s[0][j] - mx), e1 = __expf(s[1][j] - mx);
                float e2 = __expf(s[2][j] - mx), e3 = __expf(s[3][j] - mx);
                float sm = e0 + e1 + e2 + e3;
#pragma unroll
                for (int o = 1; o < 16; o <<= 1) sm += __shfl_xor(sm, o, 16);
                float inv = 1.0f / sm;
                s[0][j] = e0 * inv; s[1][j] = e1 * inv;
                s[2][j] = e2 * inv; s[3][j] = e3 * inv;
            }
#pragma unroll
            for (int nt = 0; nt < 4; ++nt)
#pragma unroll
                for (int j = 0; j < 8; ++j)
                    sPh[(mt * 16 + j + rhalf) * 64 + nt * 16 + ncol] =
                        (_Float16)s[nt][j];
        }
        // context = P @ V (K = 64 -> 2 wmma steps)
        for (int mt = 0; mt < 4; ++mt) {
#pragma unroll
            for (int dt = 0; dt < 2; ++dt) {
                f8 acc = zero8();
#pragma unroll
                for (int kk = 0; kk < 2; ++kk) {
                    h16 pa = load_a_lds(sPh, 64, mt * 16, kk * 32, lane);
                    h16 vb = load_b_lds(sVt, 64, h * 32 + dt * 16, kk * 32, lane);
                    acc = WMMA_F16(pa, vb, acc);
                }
                int col = h * 32 + dt * 16 + ncol;
#pragma unroll
                for (int j = 0; j < 8; ++j)
                    sXO[(mt * 16 + j + rhalf) * 192 + col] = (_Float16)acc[j];
            }
        }
    }
    __syncthreads();

    // ---------------- Phase D: output projection y = ctx @ Wproj^T + b ----------------
    float* ow = out + (size_t)w * 49 * 192;
    for (int ct = wv; ct < 12; ct += 8) {
        f8 acc[4];
#pragma unroll
        for (int mt = 0; mt < 4; ++mt) acc[mt] = zero8();
        for (int kk = 0; kk < 6; ++kk) {
            h16 bt;
            if constexpr (WF16) bt = load_b_gf16(wprojh, 192, ct * 16, kk * 32, lane);
            else                bt = load_b_gf32(wproj,  192, ct * 16, kk * 32, lane);
#pragma unroll
            for (int mt = 0; mt < 4; ++mt) {
                h16 at = load_a_lds(sXO, 192, mt * 16, kk * 32, lane);
                acc[mt] = WMMA_F16(at, bt, acc[mt]);
            }
        }
        const int colg  = ct * 16 + ncol;
        const float bias = bproj[colg];
#pragma unroll
        for (int mt = 0; mt < 4; ++mt)
#pragma unroll
            for (int j = 0; j < 8; ++j) {
                int m = mt * 16 + j + rhalf;
                if (m < 49) ow[m * 192 + colg] = acc[mt][j] + bias;
            }
    }
}

extern "C" void kernel_launch(void* const* d_in, const int* in_sizes, int n_in,
                              void* d_out, int out_size, void* d_ws, size_t ws_size,
                              hipStream_t stream) {
    const float* x     = (const float*)d_in[0];
    const float* qg    = (const float*)d_in[1];
    const float* wqkv  = (const float*)d_in[2];
    const float* bqkv  = (const float*)d_in[3];
    const float* btab  = (const float*)d_in[4];
    const float* wproj = (const float*)d_in[5];
    const float* bproj = (const float*)d_in[6];
    float* out = (float*)d_out;
    (void)in_sizes; (void)n_in; (void)out_size;

    const size_t QH = 301056, WQH = 73728, WPH = 36864;
    const size_t need = (QH + 512 + WQH + WPH) * sizeof(_Float16);
    _Float16* qh     = (_Float16*)d_ws;
    _Float16* wqkvh  = qh + QH + 512;
    _Float16* wprojh = wqkvh + WQH;

    dim3 grid(2048), block(256);
    if (d_ws != nullptr && ws_size >= need) {
        hipLaunchKernelGGL(wga_prep, dim3((301056 + 255) / 256), block, 0, stream,
                           qg, wqkv, wproj, qh, wqkvh, wprojh);
        hipLaunchKernelGGL((wga_fused<true>), grid, block, 0, stream,
                           x, qg, qh, wqkv, wqkvh, bqkv, btab, wproj, wprojh, bproj, out);
    } else {
        hipLaunchKernelGGL((wga_fused<false>), grid, block, 0, stream,
                           x, qg, qh, wqkv, wqkvh, bqkv, btab, wproj, wprojh, bproj, out);
    }
}
